// ScratchMPNN_50611894616079
// MI455X (gfx1250) — matrix operand order repaired and checked by reading, not compiled, not run on hardware
//
#include <hip/hip_runtime.h>
#include <hip/hip_bf16.h>

#define N_NODES   50000
#define N_EDGES   800000
#define IN_CH     256
#define HID       128
#define EDGE_DIM  64
#define N_CLASSES 10
#define N_GRAPHS  512

typedef __attribute__((ext_vector_type(16))) __bf16       v16bf;
typedef __attribute__((ext_vector_type(2)))  __bf16       bf16x2;
typedef __attribute__((ext_vector_type(8)))  float        v8f;
typedef __attribute__((ext_vector_type(4)))  unsigned int v4u;
typedef __attribute__((ext_vector_type(8)))  int          v8i;
typedef __attribute__((ext_vector_type(4)))  int          v4i;

#if defined(__has_builtin)
#  if __has_builtin(__builtin_amdgcn_tensor_load_to_lds) && __has_builtin(__builtin_amdgcn_s_wait_tensorcnt)
#    define USE_TDM 1
#  endif
#endif
#ifndef USE_TDM
#  define USE_TDM 0
#endif

__device__ __forceinline__ void atomAddF32(float* p, float v) {
    unsafeAtomicAdd(p, v);   // GLOBAL_ATOMIC_ADD_F32, L2-resident target
}

__device__ __forceinline__ unsigned packbf2(float a, float b) {
    bf16x2 v;
    v[0] = (__bf16)a;
    v[1] = (__bf16)b;
    return __builtin_bit_cast(unsigned, v);
}

#if USE_TDM
// Issue a 1-D contiguous TDM copy: `elems` f32 from `src` -> LDS offset `ldsOff`.
// Descriptor packing per CDNA5 ISA (D# groups 0/1).
// This toolchain exposes the 6-arg builtin: (v4u, v8i, v4i, v4i, v8i, i32 cpol).
__device__ __forceinline__ void tdm_copy_f32(const float* src, unsigned ldsOff, unsigned elems) {
    unsigned long long ga = (unsigned long long)src;
    v4u g0;
    g0[0] = 1u;                                           // count=1, user mode
    g0[1] = ldsOff;                                       // lds_addr [63:32]
    g0[2] = (unsigned)ga;                                 // global_addr [95:64]
    g0[3] = ((unsigned)(ga >> 32) & 0x01FFFFFFu) | (2u << 30); // addr[56:32] | type=2
    v8i g1;
    g1[0] = (int)(2u << 16);                              // data_size = 4 bytes
    g1[1] = (int)((elems & 0xFFFFu) << 16);               // tensor_dim0 [15:0]
    g1[2] = (int)((elems >> 16) | (1u << 16));            // tensor_dim0 [31:16] | tensor_dim1=1
    g1[3] = (int)(elems << 16);                           // tile_dim0 = elems (<=4096 fits 16b)
    g1[4] = 0;                                            // tile_dim1=0, tile_dim2=0 (1-D)
    g1[5] = (int)elems;                                   // tensor_dim0_stride (unused, benign)
    g1[6] = 0;
    g1[7] = 0;
    v4i g2 = {0, 0, 0, 0};                                // <=2-D: groups 2/3 unused
    v4i g3 = {0, 0, 0, 0};
    v8i g4 = {0, 0, 0, 0, 0, 0, 0, 0};
    __builtin_amdgcn_tensor_load_to_lds(g0, g1, g2, g3, g4, 0);
}
#endif

// ---------------------------------------------------------------------------
// Utility kernels
// ---------------------------------------------------------------------------
__global__ void zero_f32(float* __restrict__ p, int n) {
    int i = blockIdx.x * blockDim.x + threadIdx.x;
    int stride = gridDim.x * blockDim.x;
    for (; i < n; i += stride) p[i] = 0.0f;
}

__global__ void relu_ip(float* __restrict__ p, int n) {
    int i = blockIdx.x * blockDim.x + threadIdx.x;
    int stride = gridDim.x * blockDim.x;
    for (; i < n; i += stride) p[i] = fmaxf(p[i], 0.0f);
}

// Pack W [K x HID] f32 into per-lane WMMA B fragments (bf16), so a wave can
// fetch its 32-byte operand with two global_load_b128.
// B-layout (16-bit B, 32x16): lanes 0-15 hold K 0..7 & 16..23, lanes 16-31 K 8..15 & 24..31.
// Fragment index: [ntile][ktile][lane][e], e = (k&7) + 8*((k&31)>>4), lane = (n&15) + 16*((k&31)>>3 & 1).
__global__ void pack_W(const float* __restrict__ W, unsigned* __restrict__ Wp, int K) {
    int p = blockIdx.x * blockDim.x + threadIdx.x;      // one thread per bf16 pair
    int total = (K / 2) * HID;
    if (p >= total) return;
    int k = 2 * (p / HID);
    int n = p % HID;
    float v0 = W[(size_t)k * HID + n];
    float v1 = W[(size_t)(k + 1) * HID + n];
    int nt = n >> 4, nn = n & 15, kk = k & 31, t = k >> 5;
    int ln = nn + (((kk >> 3) & 1) << 4);
    int e  = (kk & 7) + ((kk >> 4) << 3);               // even
    Wp[(((size_t)nt * (K / 32) + t) * 32 + ln) * 8 + (e >> 1)] = packbf2(v0, v1);
}

// ---------------------------------------------------------------------------
// Fragment helpers: A-layout (16-bit A, 16x32): lanes 0-15: K 0..7 & 16..23;
// lanes 16-31: K 8..15 & 24..31.  Fragment: [ktile][lane][e],
// lane = m + 16*((k&31)>>3 & 1), e = (k&7) + 8*((k&31)>>4).
// ---------------------------------------------------------------------------
__device__ __forceinline__ void store_A_pair(unsigned* frag, int m, int k, float a0, float a1) {
    int t = k >> 5, kk = k & 31;
    int ln = m + (((kk >> 3) & 1) << 4);
    int e  = (kk & 7) + ((kk >> 4) << 3);               // even (k even)
    frag[(t * 32 + ln) * 8 + (e >> 1)] = packbf2(a0, a1);
}

// ---------------------------------------------------------------------------
// Dense node GEMM:  out[M x HID] = A[M x K] @ W + bias
// Block = 256 thr = 8 waves; block owns a 16-row tile (contiguous in memory
// -> one 1-D TDM transfer), wave w owns column tile [16w,16w+16).
// ---------------------------------------------------------------------------
template <int K>
__global__ __launch_bounds__(256)
void node_gemm(const float* __restrict__ A, const unsigned* __restrict__ Wp,
               const float* __restrict__ bias, float* __restrict__ out) {
    constexpr int KT = K / 32;
    __shared__ __align__(32) unsigned Afrag[KT * 32 * 8];   // bf16 fragments
    const int rowbase = blockIdx.x * 16;
    const int tid = threadIdx.x;

#if USE_TDM
    __shared__ __align__(16) float Af[16 * K];              // raw f32 tile (TDM dest)
    if (tid < 32) {   // wave 0 issues the DMA and waits on its TENSORcnt
        tdm_copy_f32(A + (size_t)rowbase * K, (unsigned)(unsigned long long)(void*)Af, 16 * K);
        __builtin_amdgcn_s_wait_tensorcnt(0);
    }
    __syncthreads();
    for (int p = tid; p < 8 * K; p += 256) {                // f32 LDS -> bf16 fragments
        int m = p / (K / 2), k = 2 * (p % (K / 2));
        store_A_pair(Afrag, m, k, Af[m * K + k], Af[m * K + k + 1]);
    }
#else
    for (int p = tid; p < 8 * K; p += 256) {                // global -> bf16 fragments
        int m = p / (K / 2), k = 2 * (p % (K / 2));
        const float* row = A + (size_t)(rowbase + m) * K;
        store_A_pair(Afrag, m, k, row[k], row[k + 1]);
    }
#endif
    __syncthreads();

    const int lane = tid & 31, wave = tid >> 5;
    const int nb = wave * 16, m = lane & 15;
    v8f acc = {};
#pragma unroll
    for (int t = 0; t < KT; ++t) {
        v16bf a = *(const v16bf*)&Afrag[(t * 32 + lane) * 8];                     // 2x ds_load_b128
        v16bf b = *(const v16bf*)&Wp[(((size_t)wave * KT + t) * 32 + lane) * 8];  // 2x global_load_b128
        acc = __builtin_amdgcn_wmma_f32_16x16x32_bf16(false, a, false, b, (short)0, acc, false, false);
    }

    const float bv = bias[nb + m];
    const int mrow = (lane >> 4) << 3;
#pragma unroll
    for (int r = 0; r < 8; ++r)
        out[(size_t)(rowbase + mrow + r) * HID + nb + m] = acc[r] + bv;
}

// ---------------------------------------------------------------------------
// Fused edge layer: msg = edge_attr_tile @ We + be + h_pre[col]; aggr[row] += msg
// edge_attr tile = 16 x 64 f32 = one contiguous 4KB span -> 1-D TDM transfer.
// ---------------------------------------------------------------------------
__global__ __launch_bounds__(256)
void edge_mp(const float* __restrict__ EA, const int* __restrict__ erow,
             const int* __restrict__ ecol, const float* __restrict__ hpre,
             const unsigned* __restrict__ Wep, const float* __restrict__ be,
             float* __restrict__ aggr) {
    constexpr int KT = EDGE_DIM / 32;                       // 2
    __shared__ __align__(32) unsigned Efrag[KT * 32 * 8];
    __shared__ int r16[16], c16[16];
    const int ebase = blockIdx.x * 16;
    const int tid = threadIdx.x;

    if (tid >= 64 && tid < 80)       r16[tid - 64] = erow[ebase + tid - 64];
    else if (tid >= 80 && tid < 96)  c16[tid - 80] = ecol[ebase + tid - 80];

#if USE_TDM
    __shared__ __align__(16) float Ef[16 * EDGE_DIM];
    if (tid < 32) {
        tdm_copy_f32(EA + (size_t)ebase * EDGE_DIM, (unsigned)(unsigned long long)(void*)Ef,
                     16 * EDGE_DIM);
        __builtin_amdgcn_s_wait_tensorcnt(0);
    }
    __syncthreads();
    for (int p = tid; p < 8 * EDGE_DIM; p += 256) {
        int m = p / (EDGE_DIM / 2), k = 2 * (p % (EDGE_DIM / 2));
        store_A_pair(Efrag, m, k, Ef[m * EDGE_DIM + k], Ef[m * EDGE_DIM + k + 1]);
    }
#else
    for (int p = tid; p < 8 * EDGE_DIM; p += 256) {
        int m = p / (EDGE_DIM / 2), k = 2 * (p % (EDGE_DIM / 2));
        const float* row = EA + (size_t)(ebase + m) * EDGE_DIM;
        store_A_pair(Efrag, m, k, row[k], row[k + 1]);
    }
#endif
    __syncthreads();

    const int lane = tid & 31, wave = tid >> 5;
    const int nb = wave * 16, m = lane & 15;
    v8f acc = {};
#pragma unroll
    for (int t = 0; t < KT; ++t) {
        v16bf a = *(const v16bf*)&Efrag[(t * 32 + lane) * 8];
        v16bf b = *(const v16bf*)&Wep[(((size_t)wave * KT + t) * 32 + lane) * 8];
        acc = __builtin_amdgcn_wmma_f32_16x16x32_bf16(false, a, false, b, (short)0, acc, false, false);
    }

    const int f = nb + m;
    const float bv = be[f];
    const int mrow = (lane >> 4) << 3;
#pragma unroll
    for (int r = 0; r < 8; ++r) {
        int e = mrow + r;
        float val = acc[r] + bv + hpre[(size_t)c16[e] * HID + f];   // coalesced L2 gather
        atomAddF32(&aggr[(size_t)r16[e] * HID + f], val);           // scatter_add
    }
}

// ---------------------------------------------------------------------------
// Mean pool + classifier
// ---------------------------------------------------------------------------
__global__ void pool_sum(const float* __restrict__ h, const int* __restrict__ batch,
                         float* __restrict__ pooled, float* __restrict__ counts) {
    int n = blockIdx.x;
    int f = threadIdx.x;
    int g = batch[n];
    atomAddF32(&pooled[g * HID + f], h[(size_t)n * HID + f]);
    if (f == 0) atomAddF32(&counts[g], 1.0f);
}

__global__ void classify(const float* __restrict__ pooled, const float* __restrict__ counts,
                         const float* __restrict__ Wc, const float* __restrict__ bc,
                         float* __restrict__ out) {
    int g = blockIdx.x;
    int c = threadIdx.x;
    if (c >= N_CLASSES) return;
    float inv = 1.0f / fmaxf(counts[g], 1.0f);
    float s = 0.0f;
    for (int f = 0; f < HID; ++f)
        s += pooled[g * HID + f] * inv * Wc[f * N_CLASSES + c];
    out[g * N_CLASSES + c] = s + bc[c];
}

// ---------------------------------------------------------------------------
extern "C" void kernel_launch(void* const* d_in, const int* in_sizes, int n_in,
                              void* d_out, int out_size, void* d_ws, size_t ws_size,
                              hipStream_t stream) {
    const float* x    = (const float*)d_in[0];
    const int*   eidx = (const int*)  d_in[1];   // per harness contract: int32
    const float* ea   = (const float*)d_in[2];
    const int*   bat  = (const int*)  d_in[3];
    const float* W1   = (const float*)d_in[4];   const float* b1  = (const float*)d_in[5];
    const float* We1  = (const float*)d_in[6];   const float* be1 = (const float*)d_in[7];
    const float* W2   = (const float*)d_in[8];   const float* b2  = (const float*)d_in[9];
    const float* We2  = (const float*)d_in[10];  const float* be2 = (const float*)d_in[11];
    const float* Wc   = (const float*)d_in[12];  const float* bc  = (const float*)d_in[13];
    const int* erow = eidx;
    const int* ecol = eidx + N_EDGES;

    char* ws = (char*)d_ws;
    size_t off = 0;
    auto carve = [&](size_t bytes) {
        void* p = ws + off;
        off = (off + bytes + 255) & ~(size_t)255;
        return p;
    };
    float*    B0     = (float*)   carve(sizeof(float) * (size_t)N_NODES * HID);
    float*    B1     = (float*)   carve(sizeof(float) * (size_t)N_NODES * HID);
    unsigned* W1p    = (unsigned*)carve(sizeof(__bf16) * IN_CH    * HID);
    unsigned* We1p   = (unsigned*)carve(sizeof(__bf16) * EDGE_DIM * HID);
    unsigned* W2p    = (unsigned*)carve(sizeof(__bf16) * HID      * HID);
    unsigned* We2p   = (unsigned*)carve(sizeof(__bf16) * EDGE_DIM * HID);
    float*    pooled = (float*)   carve(sizeof(float) * N_GRAPHS * HID);
    float*    counts = (float*)   carve(sizeof(float) * N_GRAPHS);

    const int NH = N_NODES * HID;

    // Weights -> bf16 WMMA fragments (once per call; deterministic)
    pack_W<<<(IN_CH*HID/2    + 255)/256, 256, 0, stream>>>(W1,  W1p,  IN_CH);
    pack_W<<<(EDGE_DIM*HID/2 + 255)/256, 256, 0, stream>>>(We1, We1p, EDGE_DIM);
    pack_W<<<(HID*HID/2      + 255)/256, 256, 0, stream>>>(W2,  W2p,  HID);
    pack_W<<<(EDGE_DIM*HID/2 + 255)/256, 256, 0, stream>>>(We2, We2p, EDGE_DIM);

    // Layer 1
    zero_f32<<<2048, 256, 0, stream>>>(B1, NH);
    node_gemm<IN_CH><<<N_NODES/16, 256, 0, stream>>>(x, W1p, b1, B0);
    edge_mp<<<N_EDGES/16, 256, 0, stream>>>(ea, erow, ecol, B0, We1p, be1, B1);
    relu_ip<<<2048, 256, 0, stream>>>(B1, NH);

    // Layer 2 (B1 = h1 consumed by node_gemm, then recycled as aggr2)
    node_gemm<HID><<<N_NODES/16, 256, 0, stream>>>(B1, W2p, b2, B0);
    zero_f32<<<2048, 256, 0, stream>>>(B1, NH);
    edge_mp<<<N_EDGES/16, 256, 0, stream>>>(ea, erow, ecol, B0, We2p, be2, B1);
    relu_ip<<<2048, 256, 0, stream>>>(B1, NH);

    // Mean pool + classifier
    zero_f32<<<(N_GRAPHS*HID + 255)/256, 256, 0, stream>>>(pooled, N_GRAPHS*HID);
    zero_f32<<<2, 256, 0, stream>>>(counts, N_GRAPHS);
    pool_sum<<<N_NODES, HID, 0, stream>>>(B1, bat, pooled, counts);
    classify<<<N_GRAPHS, 32, 0, stream>>>(pooled, counts, Wc, bc, (float*)d_out);
}
